// GptOssExperts_90778428768673
// MI455X (gfx1250) — compile-verified, matrix operand order checked by reading, not run on hardware
//
#include <hip/hip_runtime.h>
#include <math.h>

// ---------------------------------------------------------------------------
// GPT-OSS MoE forward for MI455X (gfx1250, wave32, WMMA).
//   GEMM1+SwiGLU fused, GEMM2+weighted-combine fused, bf16 WMMA w/ fp32 acc.
//   Double-buffered LDS tiles: one barrier per K-step, global fill of step
//   k+1 overlaps the v_wmma issue of step k.
// Workspace layout (requires ws_size >= ~161 MB):
//   [ctrl 16KB][p_token 128KB][p_weight 128KB][hidden_bf16 32MB][inter_bf16 128MB]
// ---------------------------------------------------------------------------

#define TT    16384
#define HH    1024
#define EE    16
#define FF    2048
#define TWO_F 4096
#define TK    32768          // T * K (K = 2)
#define MAX_TILES 272        // sum ceil(cnt_e/128) <= 32768/128 + 16

#define TILE_USH 7680        // per-buffer LDS ushorts: As 128*40 + Bs 64*40
#define AS_USH   5120        // 128 * 40

typedef __attribute__((ext_vector_type(16))) __bf16          v16bf;
typedef __attribute__((ext_vector_type(8)))  float           v8f;
typedef __attribute__((ext_vector_type(8)))  unsigned short  us8;
typedef __attribute__((ext_vector_type(4)))  float           f4;

union FragAB { us8 h[2]; v16bf v; };

// ctrl[] int offsets
#define C_COUNT 0
#define C_OFFS  16
#define C_CURS  33
#define C_NUMT  49
#define C_TE    64
#define C_TR    576

__device__ __forceinline__ unsigned short f2bf(float x) {
  unsigned int u = __float_as_uint(x);
  u += 0x7FFFu + ((u >> 16) & 1u);       // round-to-nearest-even
  return (unsigned short)(u >> 16);
}

// -------------------------- routing kernels --------------------------------

__global__ void k_init(int* ctrl) {
  int i = threadIdx.x;
  if (i < EE) ctrl[C_COUNT + i] = 0;
  if (i == EE) ctrl[C_NUMT] = 0;
}

__global__ void k_count(const int* __restrict__ ids, int* ctrl) {
  int i = blockIdx.x * 256 + threadIdx.x;
  if (i < TK) atomicAdd(&ctrl[C_COUNT + ids[i]], 1);
}

__global__ void k_scan(int* ctrl) {
  if (threadIdx.x == 0) {
    int off = 0;
    for (int e = 0; e < EE; ++e) {
      ctrl[C_OFFS + e] = off;
      ctrl[C_CURS + e] = off;
      off += ctrl[C_COUNT + e];
    }
    ctrl[C_OFFS + EE] = off;
    int nt = 0;
    for (int e = 0; e < EE; ++e)
      for (int r = ctrl[C_OFFS + e]; r < ctrl[C_OFFS + e + 1]; r += 128) {
        ctrl[C_TE + nt] = e; ctrl[C_TR + nt] = r; ++nt;
      }
    ctrl[C_NUMT] = nt;
  }
}

__global__ void k_scatter(const int* __restrict__ ids, const float* __restrict__ tw,
                          int* ctrl, int* __restrict__ ptok, float* __restrict__ pwt) {
  int i = blockIdx.x * 256 + threadIdx.x;
  if (i < TK) {
    int e = ids[i];
    int p = atomicAdd(&ctrl[C_CURS + e], 1);
    ptok[p] = i >> 1;        // token index (K = 2)
    pwt[p]  = tw[i];
  }
}

__global__ void k_zero_out(float* __restrict__ out) {
  size_t i = (size_t)(blockIdx.x * 256 + threadIdx.x) * 4;
  *(f4*)(out + i) = (f4){0.f, 0.f, 0.f, 0.f};
}

__global__ void k_cvt(const float* __restrict__ x, unsigned short* __restrict__ y) {
  size_t i = (size_t)(blockIdx.x * 256 + threadIdx.x) * 4;
  f4 v = *(const f4*)(x + i);
  y[i + 0] = f2bf(v.x); y[i + 1] = f2bf(v.y);
  y[i + 2] = f2bf(v.z); y[i + 3] = f2bf(v.w);
}

// ----------------------------- GEMM helpers --------------------------------
// LDS tiles: row stride 40 bf16 (80B) => every 8-elem chunk is 16B aligned.
// A-frag (16x32 bf16): lane<16 row m=lane, k-chunks {0..7,16..23};
//                      lane>=16 row m=lane-16, k-chunks {8..15,24..31}.
// B stored N-major in LDS so B-frags load with the same pattern.

#define WMMA_BF16(A, B, C) \
  __builtin_amdgcn_wmma_f32_16x16x32_bf16(false, (A), false, (B), (short)0, (C), false, false)

// ----------------------------- GEMM 1 --------------------------------------
// [rows,1024](bf16 gathered) x gate_up[e][1024,4096](f32->bf16) ; fused SwiGLU
__global__ __launch_bounds__(256, 2)
void k_gemm1(const unsigned short* __restrict__ hidbf,
             const float* __restrict__ w1,
             const float* __restrict__ b1,
             const int* __restrict__ ctrl,
             const int* __restrict__ ptok,
             unsigned short* __restrict__ inter) {
  __shared__ __align__(16) unsigned char smem[33792];   // max(2 tile bufs 30720, epi 33792)
  unsigned short* sus = (unsigned short*)smem;

  if ((int)blockIdx.x >= ctrl[C_NUMT]) return;
  const int e      = ctrl[C_TE + blockIdx.x];
  const int row0   = ctrl[C_TR + blockIdx.x];
  const int rowEnd = ctrl[C_OFFS + e + 1];
  const int nB     = blockIdx.y;                        // 64 cols of 2F

  const int t    = threadIdx.x;
  const int lane = t & 31, wv = t >> 5;
  const int hl   = lane >> 4, r = lane & 15;
  const int wm   = (wv & 3) * 32;                       // wave row base
  const int wnb  = (wv >> 2) * 32;                      // wave col base

  // A fill: 2 threads/row, 16 bf16 each
  const int ar = t >> 1, ac = (t & 1) << 4;
  const int pA = row0 + ar;
  const bool aval = pA < rowEnd;
  const unsigned short* asrc = aval ? (hidbf + (size_t)ptok[pA] * HH + ac) : (const unsigned short*)0;

  // B fill: thread -> (k = t>>3, n0 = (t&7)*8)
  const int bk = t >> 3, bn0 = (t & 7) << 3;
  const float* bsrc = w1 + (size_t)e * HH * TWO_F + (size_t)bk * TWO_F + (size_t)nB * 64 + bn0;

  // Invalid (past-group-end) A rows stay zero: clear both buffers once.
  if (!aval) {
    us8 z = {0, 0, 0, 0, 0, 0, 0, 0};
    us8* d0 = (us8*)(sus + 0 * TILE_USH + ar * 40 + ac);
    us8* d1 = (us8*)(sus + 1 * TILE_USH + ar * 40 + ac);
    d0[0] = z; d0[1] = z; d1[0] = z; d1[1] = z;
  }

  auto fill = [&](int kb, int buf) {
    unsigned short* A = sus + buf * TILE_USH;
    unsigned short* B = A + AS_USH;
    if (aval) {
      const us8* s = (const us8*)(asrc + kb * 32);
      us8* d = (us8*)(A + ar * 40 + ac);
      d[0] = s[0]; d[1] = s[1];
    }
    const float* bs = bsrc + (size_t)kb * 32 * TWO_F;
    f4 x0 = *(const f4*)(bs);
    f4 x1 = *(const f4*)(bs + 4);
#pragma unroll
    for (int q2 = 0; q2 < 4; ++q2) {
      B[(bn0 + q2) * 40 + bk]     = f2bf(x0[q2]);
      B[(bn0 + 4 + q2) * 40 + bk] = f2bf(x1[q2]);
    }
  };

  v8f acc[2][2] = {};

  fill(0, 0);
  __syncthreads();
  const int KB = HH / 32;
  for (int kb = 0; kb < KB; ++kb) {
    const int cur = kb & 1;
    if (kb + 1 < KB) fill(kb + 1, cur ^ 1);   // overlaps WMMAs of this step

    unsigned short* A = sus + cur * TILE_USH;
    unsigned short* B = A + AS_USH;
    FragAB a0, a1, b0, b1;
    const us8* q;
    q = (const us8*)(A + (wm + r) * 40 + (hl << 3));        a0.h[0] = q[0]; a0.h[1] = q[2];
    q = (const us8*)(A + (wm + 16 + r) * 40 + (hl << 3));   a1.h[0] = q[0]; a1.h[1] = q[2];
    q = (const us8*)(B + (wnb + r) * 40 + (hl << 3));       b0.h[0] = q[0]; b0.h[1] = q[2];
    q = (const us8*)(B + (wnb + 16 + r) * 40 + (hl << 3));  b1.h[0] = q[0]; b1.h[1] = q[2];

    acc[0][0] = WMMA_BF16(a0.v, b0.v, acc[0][0]);
    acc[0][1] = WMMA_BF16(a0.v, b1.v, acc[0][1]);
    acc[1][0] = WMMA_BF16(a1.v, b0.v, acc[1][0]);
    acc[1][1] = WMMA_BF16(a1.v, b1.v, acc[1][1]);
    __syncthreads();
  }

  // Epilogue: stage 32x32 fp32 tile per wave in LDS, pair (gate,up) columns.
  // (Trailing loop barrier already fences the LDS reuse.)
  float* Cw = (float*)smem + wv * (32 * 33);
#pragma unroll
  for (int i = 0; i < 2; ++i)
#pragma unroll
    for (int j = 0; j < 2; ++j)
#pragma unroll
      for (int v = 0; v < 8; ++v)
        Cw[(i * 16 + v + hl * 8) * 33 + j * 16 + r] = acc[i][j][v];

  const int p = row0 + wm + lane;
  if (p < rowEnd) {
    const int cg0 = nB * 64 + wnb;                      // even global col base
    const float* bias = b1 + (size_t)e * TWO_F + cg0;
    const float* Cr = Cw + lane * 33;
    unsigned short* dst = inter + (size_t)p * FF + (cg0 >> 1);
#pragma unroll
    for (int f = 0; f < 16; ++f) {
      float g = Cr[2 * f]     + bias[2 * f];
      float u = Cr[2 * f + 1] + bias[2 * f + 1];
      g = fminf(g, 7.0f);
      u = fminf(fmaxf(u, -7.0f), 7.0f);
      float glu = g / (1.0f + __expf(-1.702f * g));
      dst[f] = f2bf((u + 1.0f) * glu);
    }
  }
}

// ----------------------------- GEMM 2 --------------------------------------
// [rows,2048](bf16) x down[e][2048,1024](f32->bf16); +bias, *topk_w, atomic add
__global__ __launch_bounds__(256, 2)
void k_gemm2(const unsigned short* __restrict__ inter,
             const float* __restrict__ w2,
             const float* __restrict__ b2,
             const int* __restrict__ ctrl,
             const int* __restrict__ ptok,
             const float* __restrict__ pwt,
             float* __restrict__ out) {
  __shared__ __align__(16) unsigned char smem[30720];    // 2 tile buffers
  unsigned short* sus = (unsigned short*)smem;

  if ((int)blockIdx.x >= ctrl[C_NUMT]) return;
  const int e      = ctrl[C_TE + blockIdx.x];
  const int row0   = ctrl[C_TR + blockIdx.x];
  const int rowEnd = ctrl[C_OFFS + e + 1];
  const int nB     = blockIdx.y;                         // 64 cols of H

  const int t    = threadIdx.x;
  const int lane = t & 31, wv = t >> 5;
  const int hl   = lane >> 4, r = lane & 15;
  const int wm   = (wv & 3) * 32;
  const int wnb  = (wv >> 2) * 32;

  const int ar = t >> 1, ac = (t & 1) << 4;
  const int pA = row0 + ar;
  const bool aval = pA < rowEnd;
  const unsigned short* asrc = aval ? (inter + (size_t)pA * FF + ac) : (const unsigned short*)0;

  const int bk = t >> 3, bn0 = (t & 7) << 3;
  const float* bsrc = w2 + (size_t)e * FF * HH + (size_t)bk * HH + (size_t)nB * 64 + bn0;

  if (!aval) {
    us8 z = {0, 0, 0, 0, 0, 0, 0, 0};
    us8* d0 = (us8*)(sus + 0 * TILE_USH + ar * 40 + ac);
    us8* d1 = (us8*)(sus + 1 * TILE_USH + ar * 40 + ac);
    d0[0] = z; d0[1] = z; d1[0] = z; d1[1] = z;
  }

  auto fill = [&](int kb, int buf) {
    unsigned short* A = sus + buf * TILE_USH;
    unsigned short* B = A + AS_USH;
    if (aval) {
      const us8* s = (const us8*)(asrc + kb * 32);
      us8* d = (us8*)(A + ar * 40 + ac);
      d[0] = s[0]; d[1] = s[1];
    }
    const float* bs = bsrc + (size_t)kb * 32 * HH;
    f4 x0 = *(const f4*)(bs);
    f4 x1 = *(const f4*)(bs + 4);
#pragma unroll
    for (int q2 = 0; q2 < 4; ++q2) {
      B[(bn0 + q2) * 40 + bk]     = f2bf(x0[q2]);
      B[(bn0 + 4 + q2) * 40 + bk] = f2bf(x1[q2]);
    }
  };

  v8f acc[2][2] = {};

  fill(0, 0);
  __syncthreads();
  const int KB = FF / 32;
  for (int kb = 0; kb < KB; ++kb) {
    const int cur = kb & 1;
    if (kb + 1 < KB) fill(kb + 1, cur ^ 1);

    unsigned short* A = sus + cur * TILE_USH;
    unsigned short* B = A + AS_USH;
    FragAB a0, a1, b0, b1;
    const us8* q;
    q = (const us8*)(A + (wm + r) * 40 + (hl << 3));        a0.h[0] = q[0]; a0.h[1] = q[2];
    q = (const us8*)(A + (wm + 16 + r) * 40 + (hl << 3));   a1.h[0] = q[0]; a1.h[1] = q[2];
    q = (const us8*)(B + (wnb + r) * 40 + (hl << 3));       b0.h[0] = q[0]; b0.h[1] = q[2];
    q = (const us8*)(B + (wnb + 16 + r) * 40 + (hl << 3));  b1.h[0] = q[0]; b1.h[1] = q[2];

    acc[0][0] = WMMA_BF16(a0.v, b0.v, acc[0][0]);
    acc[0][1] = WMMA_BF16(a0.v, b1.v, acc[0][1]);
    acc[1][0] = WMMA_BF16(a1.v, b0.v, acc[1][0]);
    acc[1][1] = WMMA_BF16(a1.v, b1.v, acc[1][1]);
    __syncthreads();
  }

  // Epilogue: +down_bias, *topk_weight, atomic accumulate into output token row
#pragma unroll
  for (int i = 0; i < 2; ++i)
#pragma unroll
    for (int j = 0; j < 2; ++j) {
      const int hcol = nB * 64 + wnb + j * 16 + r;
      const float bb = b2[(size_t)e * HH + hcol];
#pragma unroll
      for (int v = 0; v < 8; ++v) {
        const int p = row0 + wm + i * 16 + hl * 8 + v;
        if (p < rowEnd) {
          const float val = acc[i][j][v] + bb;
          atomicAdd(out + (size_t)ptok[p] * HH + hcol, pwt[p] * val);
        }
      }
    }
}

// ------------------------------ launch -------------------------------------

extern "C" void kernel_launch(void* const* d_in, const int* in_sizes, int n_in,
                              void* d_out, int out_size, void* d_ws, size_t ws_size,
                              hipStream_t stream) {
  (void)in_sizes; (void)n_in; (void)out_size; (void)ws_size;
  const float* hidden = (const float*)d_in[0];
  const float* tw     = (const float*)d_in[1];
  const int*   ids    = (const int*)d_in[2];
  const float* w1     = (const float*)d_in[3];
  const float* b1     = (const float*)d_in[4];
  const float* w2     = (const float*)d_in[5];
  const float* b2     = (const float*)d_in[6];
  float* out = (float*)d_out;

  char* ws = (char*)d_ws;
  int*   ctrl = (int*)ws;                                        // 16 KB
  int*   ptok = (int*)(ws + (16 << 10));                         // 128 KB
  float* pwt  = (float*)(ws + (16 << 10) + (128 << 10));         // 128 KB
  unsigned short* hidbf = (unsigned short*)(ws + (16 << 10) + (256 << 10)); // 32 MB
  unsigned short* inter = hidbf + (size_t)TT * HH;               // 128 MB

  k_init   <<<1, 64, 0, stream>>>(ctrl);
  k_zero_out<<<TT * HH / 4 / 256, 256, 0, stream>>>(out);
  k_count  <<<TK / 256, 256, 0, stream>>>(ids, ctrl);
  k_scan   <<<1, 1, 0, stream>>>(ctrl);
  k_scatter<<<TK / 256, 256, 0, stream>>>(ids, tw, ctrl, ptok, pwt);
  k_cvt    <<<TT * HH / 4 / 256, 256, 0, stream>>>(hidden, hidbf);

  dim3 g1(MAX_TILES, TWO_F / 64);
  k_gemm1<<<g1, 256, 0, stream>>>(hidbf, w1, b1, ctrl, ptok, inter);
  dim3 g2(MAX_TILES, HH / 64);
  k_gemm2<<<g2, 256, 0, stream>>>(inter, w2, b2, ctrl, ptok, pwt, out);
}
